// DeformableTransformer_65773129171138
// MI455X (gfx1250) — compile-verified, matrix operand order checked by reading, not run on hardware
//
#include <hip/hip_runtime.h>
#include <hip/hip_bf16.h>
#include <math.h>

typedef __attribute__((ext_vector_type(16))) _Float16 v16h;
typedef __attribute__((ext_vector_type(8)))  float    v8f;

#define BS_   16
#define NQ_   900
#define D_    256
#define DFFN_ 1024
#define L_    6
#define HS_   8
#define HD_   2
#define TT_   7680
#define NT_   (BS_ * NQ_)      // 14400 tokens
#define BT_   (BS_ * TT_)      // 122880 src tokens

#define BM_ 128
#define BN_ 64
#define BK_ 32

// K index mapping for 16-bit A/B fragments (ISA 7.12.2):
// lane<16: halfs 0..7 -> K 0..7, halfs 8..15 -> K 16..23 ; lane>=16: +8
__device__ __forceinline__ int kmap(int h, int half16) {
    return ((h < 8) ? h : (h + 8)) + half16 * 8;
}

// ---------------------------------------------------------------------------
// Tiled WMMA GEMM:  C[M,N] = epi( (A (+A2)) @ B + bias )
// 128x64x32 tiles, 256 threads (8 waves, one 16-row strip each),
// double-buffered LDS. B staged fragment-major so frag reads are 2x b128.
// Epilogue configuration is compile-time (no runtime branches in hot loop).
// ---------------------------------------------------------------------------
template<bool HASA2, bool RELU, bool HASMUL, bool HASRES>
__global__ void __launch_bounds__(256)
gemm_wmma(const float* __restrict__ A, const float* __restrict__ A2,
          const float* __restrict__ B, const float* __restrict__ bias,
          const float* __restrict__ mulp, const float* __restrict__ resp,
          float* __restrict__ C, int M, int N, int K)
{
    __shared__ _Float16 As[2][BM_][BK_ + 8];     // row-major A tiles
    __shared__ _Float16 BsF[2][4][32][16];       // frag-major: [nsub][lane][half]

    const int t      = threadIdx.x;
    const int lane   = t & 31;
    const int wv     = t >> 5;
    const int half16 = lane >> 4;
    const int l15    = lane & 15;
    const int m0     = blockIdx.y * BM_;
    const int n0     = blockIdx.x * BN_;

    const int arow = t >> 1, acb = (t & 1) * 16;    // A tile: 128 rows x 32
    const int aclmp = min(m0 + arow, M - 1);        // clamp for partial M tile
    const int brow = t >> 3, bcb = (t & 7) * 8;     // B tile: 32 rows x 64

    v8f c0 = {}, c1 = {}, c2 = {}, c3 = {};

    const int nk = K / BK_;

    // ---- staging helper ----
    auto stage = [&](int k0, int buf) {
        // A tile (fp32 -> f16), optionally fused A+A2
        const float* ap = A + (size_t)aclmp * K + k0 + acb;
        if (k0 + BK_ < K) __builtin_prefetch(ap + BK_, 0, 1);   // global_prefetch_b8
        #pragma unroll
        for (int e = 0; e < 16; e += 4) {
            float4 va = *reinterpret_cast<const float4*>(ap + e);
            if (HASA2) {
                const float* ap2 = A2 + (size_t)aclmp * K + k0 + acb;
                float4 vb = *reinterpret_cast<const float4*>(ap2 + e);
                va.x += vb.x; va.y += vb.y; va.z += vb.z; va.w += vb.w;
            }
            As[buf][arow][acb + e + 0] = (_Float16)va.x;
            As[buf][arow][acb + e + 1] = (_Float16)va.y;
            As[buf][arow][acb + e + 2] = (_Float16)va.z;
            As[buf][arow][acb + e + 3] = (_Float16)va.w;
        }
        // B tile -> fragment-major layout (packed u16 scatter stores)
        {
            const int nbase = n0 + bcb;
            const bool okb = nbase < N;             // 8-col chunk fully in or out
            const float* bp = B + (size_t)(k0 + brow) * N + nbase;
            if (okb && k0 + BK_ < K) __builtin_prefetch(bp + (size_t)BK_ * N, 0, 1);
            float4 z4 = make_float4(0.f, 0.f, 0.f, 0.f);
            float4 v0 = okb ? *reinterpret_cast<const float4*>(bp)     : z4;
            float4 v1 = okb ? *reinterpret_cast<const float4*>(bp + 4) : z4;
            float vv[8] = {v0.x, v0.y, v0.z, v0.w, v1.x, v1.y, v1.z, v1.w};
            const int kk   = brow;
            const int ltk  = 16 * ((kk >> 3) & 1);
            const int hh   = (kk & 7) + ((kk >= 16) ? 8 : 0);
            #pragma unroll
            for (int e = 0; e < 8; ++e) {
                int nn = bcb + e;                   // local col 0..63
                BsF[buf][nn >> 4][(nn & 15) + ltk][hh] = (_Float16)vv[e];
            }
        }
    };

    stage(0, 0);

    for (int kt = 0; kt < nk; ++kt) {
        __syncthreads();
        if (kt + 1 < nk) stage((kt + 1) * BK_, (kt + 1) & 1);
        const int buf = kt & 1;

        v16h af;
        #pragma unroll
        for (int h = 0; h < 16; ++h) af[h] = As[buf][wv * 16 + l15][kmap(h, half16)];

        #pragma unroll
        for (int j = 0; j < 4; ++j) {
            v16h bf;
            #pragma unroll
            for (int h = 0; h < 16; ++h) bf[h] = BsF[buf][j][lane][h];   // 2x b128
            v8f& cc = (j == 0) ? c0 : (j == 1) ? c1 : (j == 2) ? c2 : c3;
            cc = __builtin_amdgcn_wmma_f32_16x16x32_f16(
                     false, af, false, bf, (short)0, cc, false, false);
        }
    }

    // ---- epilogue (branch-free per template config) ----
    #pragma unroll
    for (int j = 0; j < 4; ++j) {
        const v8f& cc = (j == 0) ? c0 : (j == 1) ? c1 : (j == 2) ? c2 : c3;
        int n = n0 + j * 16 + l15;
        if (n < N) {
            float bv = bias[n];
            #pragma unroll
            for (int r = 0; r < 8; ++r) {
                int m = m0 + wv * 16 + half16 * 8 + r;
                if (m < M) {
                    float v = cc[r] + bv;
                    if (RELU) v = fmaxf(v, 0.0f);
                    size_t o = (size_t)m * N + n;
                    if (HASMUL) v *= mulp[o];
                    if (HASRES) v += resp[o];
                    C[o] = v;
                }
            }
        }
    }
}

// ---------------------------------------------------------------------------
// Flash MHA: one wave per (b, h, 16-query tile). dh = 32 == WMMA K.
// K/V staged in fragment-major layout (vectorized ds_load_b128 frag reads).
// ---------------------------------------------------------------------------
__global__ void __launch_bounds__(32)
mha_flash(const float* __restrict__ Qg, const float* __restrict__ Kg,
          const float* __restrict__ Vg, float* __restrict__ Og)
{
    __shared__ _Float16 Qs[16][32];
    __shared__ _Float16 KsF[2][32][16];   // B frags of K^T: [key-sub][lane][half]
    __shared__ _Float16 VsF[2][32][16];   // B frags of V:   [dh-sub][lane][half]
    __shared__ _Float16 Ps[16][40];

    const int lane   = threadIdx.x;
    const int l15    = lane & 15;
    const int half16 = lane >> 4;
    const int qt     = blockIdx.x;
    const int b      = blockIdx.y >> 3;   // HS_ = 8
    const int h      = blockIdx.y & 7;
    const float scale = 0.17677669529663689f;   // 1/sqrt(32)

    // load Q tile (scaled), zero-pad invalid query rows
    {
        int row = lane >> 1, cb = (lane & 1) * 16;
        int q = qt * 16 + row;
        const float* qp = Qg + ((size_t)(b * NQ_ + (q < NQ_ ? q : 0))) * D_ + h * 32 + cb;
        #pragma unroll
        for (int e = 0; e < 16; ++e)
            Qs[row][cb + e] = (q < NQ_) ? (_Float16)(qp[e] * scale) : (_Float16)0.0f;
    }
    __syncthreads();

    v16h aq;
    #pragma unroll
    for (int hh = 0; hh < 16; ++hh) aq[hh] = Qs[l15][kmap(hh, half16)];

    float rowmax[8], rowsum[8];
    #pragma unroll
    for (int r = 0; r < 8; ++r) { rowmax[r] = -1e30f; rowsum[r] = 0.0f; }
    v8f o0 = {}, o1 = {};

    for (int kc = 0; kc < 29; ++kc) {        // ceil(900/32)
        const int kb = kc * 32;
        {   // lane stages one key row into fragment-major K^T and V tiles
            const int keyl = lane;
            const int key  = kb + keyl;
            const bool ok  = key < NQ_;
            const size_t base = ((size_t)(b * NQ_ + (ok ? key : 0))) * D_ + h * 32;
            const float* kp = Kg + base;
            const float* vp = Vg + base;
            const int vls = 16 * ((keyl >> 3) & 1);
            const int vh  = (keyl & 7) + ((keyl >= 16) ? 8 : 0);
            float4 z4 = make_float4(0.f, 0.f, 0.f, 0.f);
            #pragma unroll
            for (int e = 0; e < 32; e += 4) {
                float4 kv = ok ? *reinterpret_cast<const float4*>(kp + e) : z4;
                float4 vv = ok ? *reinterpret_cast<const float4*>(vp + e) : z4;
                float ka[4] = {kv.x, kv.y, kv.z, kv.w};
                float va[4] = {vv.x, vv.y, vv.z, vv.w};
                #pragma unroll
                for (int q4 = 0; q4 < 4; ++q4) {
                    int dh = e + q4;
                    KsF[keyl >> 4][(keyl & 15) + 16 * ((dh >> 3) & 1)]
                       [(dh & 7) + ((dh >= 16) ? 8 : 0)] = (_Float16)ka[q4];
                    VsF[dh >> 4][(dh & 15) + vls][vh] = (_Float16)va[q4];
                }
            }
        }
        __syncthreads();

        v16h bk0, bk1;
        #pragma unroll
        for (int hh = 0; hh < 16; ++hh) {
            bk0[hh] = KsF[0][lane][hh];
            bk1[hh] = KsF[1][lane][hh];
        }
        v8f z = {};
        v8f s0 = __builtin_amdgcn_wmma_f32_16x16x32_f16(false, aq, false, bk0, (short)0, z, false, false);
        v8f s1 = __builtin_amdgcn_wmma_f32_16x16x32_f16(false, aq, false, bk1, (short)0, z, false, false);

        if (kb + 32 > NQ_) {   // mask padded keys in last chunk
            bool bad0 = (kb + l15)      >= NQ_;
            bool bad1 = (kb + 16 + l15) >= NQ_;
            #pragma unroll
            for (int r = 0; r < 8; ++r) {
                if (bad0) s0[r] = -1e30f;
                if (bad1) s1[r] = -1e30f;
            }
        }

        // online softmax (rows live in half-waves of 16 lanes)
        #pragma unroll
        for (int r = 0; r < 8; ++r) {
            float mv = fmaxf(s0[r], s1[r]);
            mv = fmaxf(mv, __shfl_xor(mv, 1, 16));
            mv = fmaxf(mv, __shfl_xor(mv, 2, 16));
            mv = fmaxf(mv, __shfl_xor(mv, 4, 16));
            mv = fmaxf(mv, __shfl_xor(mv, 8, 16));
            float mnew = fmaxf(rowmax[r], mv);
            float sc = __expf(rowmax[r] - mnew);
            rowmax[r] = mnew;
            float p0 = __expf(s0[r] - mnew);
            float p1 = __expf(s1[r] - mnew);
            float ps = p0 + p1;
            ps += __shfl_xor(ps, 1, 16);
            ps += __shfl_xor(ps, 2, 16);
            ps += __shfl_xor(ps, 4, 16);
            ps += __shfl_xor(ps, 8, 16);
            rowsum[r] = rowsum[r] * sc + ps;
            o0[r] *= sc; o1[r] *= sc;
            int m = r + half16 * 8;            // C-layout row
            Ps[m][l15]      = (_Float16)p0;
            Ps[m][l15 + 16] = (_Float16)p1;
        }
        __syncthreads();

        // O += P @ V
        v16h ap, bv0, bv1;
        #pragma unroll
        for (int hh = 0; hh < 16; ++hh) {
            ap[hh]  = Ps[l15][kmap(hh, half16)];
            bv0[hh] = VsF[0][lane][hh];
            bv1[hh] = VsF[1][lane][hh];
        }
        o0 = __builtin_amdgcn_wmma_f32_16x16x32_f16(false, ap, false, bv0, (short)0, o0, false, false);
        o1 = __builtin_amdgcn_wmma_f32_16x16x32_f16(false, ap, false, bv1, (short)0, o1, false, false);
        __syncthreads();
    }

    #pragma unroll
    for (int r = 0; r < 8; ++r) {
        float inv = 1.0f / rowsum[r];
        int m = r + half16 * 8;
        int q = qt * 16 + m;
        if (q < NQ_) {
            size_t base = ((size_t)(b * NQ_ + q)) * D_ + h * 32;
            Og[base + l15]      = o0[r] * inv;
            Og[base + 16 + l15] = o1[r] * inv;
        }
    }
}

// ---------------------------------------------------------------------------
// LayerNorm over D=256, in place. One block per token row.
// ---------------------------------------------------------------------------
__global__ void __launch_bounds__(256)
ln_kernel(float* __restrict__ x, const float* __restrict__ g, const float* __restrict__ b)
{
    __shared__ float red[256];
    const int row = blockIdx.x, t = threadIdx.x;
    float v = x[(size_t)row * 256 + t];
    red[t] = v; __syncthreads();
    for (int s = 128; s > 0; s >>= 1) { if (t < s) red[t] += red[t + s]; __syncthreads(); }
    float mean = red[0] * (1.0f / 256.0f);
    __syncthreads();
    float d = v - mean;
    red[t] = d * d; __syncthreads();
    for (int s = 128; s > 0; s >>= 1) { if (t < s) red[t] += red[t + s]; __syncthreads(); }
    float var = red[0] * (1.0f / 256.0f);
    x[(size_t)row * 256 + t] = d * rsqrtf(var + 1e-5f) * g[t] + b[t];
}

// ---------------------------------------------------------------------------
// Precompute: sine pos-embed PE[NT,512] and rpi[NT,LV,2] (layer-invariant).
// ---------------------------------------------------------------------------
__global__ void __launch_bounds__(256)
prep_kernel(const float* __restrict__ seg, const float* __restrict__ dur,
            const float* __restrict__ vr, float* __restrict__ PE,
            float* __restrict__ rpi)
{
    const int idx = blockIdx.x;            // b*NQ + q
    const int b   = idx / NQ_;
    const int t   = threadIdx.x;
    const float c   = seg[(size_t)idx * 2 + 0];
    const float wdt = __expf(seg[(size_t)idx * 2 + 1]);
    const float twopi = 6.283185307179586f;
    #pragma unroll
    for (int rep = 0; rep < 2; ++rep) {
        int j  = t + rep * 256;
        float p = (j < 256) ? c : wdt;
        int jj = j & 255;
        int k  = jj >> 1;
        float dimt = __powf(10000.0f, (float)(2 * k) * (1.0f / 256.0f));
        float a = p * twopi / dimt;
        PE[(size_t)idx * 512 + j] = (jj & 1) ? __cosf(a) : __sinf(a);
    }
    if (t < 4) {
        float s = vr[b * 4 + t] / dur[b];
        rpi[(size_t)idx * 8 + t * 2 + 0] = c * s;
        rpi[(size_t)idx * 8 + t * 2 + 1] = wdt * s;
    }
}

// ---------------------------------------------------------------------------
// Deformable sampling.
// ---------------------------------------------------------------------------
__global__ void __launch_bounds__(256)
deform_sample(const float* __restrict__ value, const float* __restrict__ offb,
              const float* __restrict__ attL, const float* __restrict__ rpi,
              float* __restrict__ samp)
{
    const int idx = blockIdx.x;            // b*NQ + q
    const int b   = idx / NQ_;
    const int t   = threadIdx.x;
    const int h   = t >> 7;                // head (0..1)
    const int ch  = t & 127;               // channel within head
    const int lens[4]   = {4096, 2048, 1024, 512};
    const int starts[4] = {0, 4096, 6144, 7168};

    const float* lg = attL + (size_t)idx * 32 + h * 16;
    float mx = -1e30f;
    #pragma unroll
    for (int j = 0; j < 16; ++j) mx = fmaxf(mx, lg[j]);
    float attp[16], sum = 0.0f;
    #pragma unroll
    for (int j = 0; j < 16; ++j) { attp[j] = __expf(lg[j] - mx); sum += attp[j]; }
    const float inv = 1.0f / sum;

    const float* ob = offb + (size_t)idx * 32 + h * 16;
    const float* rp = rpi + (size_t)idx * 8;

    float acc = 0.0f;
    #pragma unroll
    for (int l = 0; l < 4; ++l) {
        float rc = rp[l * 2], rw = rp[l * 2 + 1];
        int Tl = lens[l];
        const float* vb = value + ((size_t)b * TT_ + starts[l]) * 256 + h * 128 + ch;
        #pragma unroll
        for (int p = 0; p < 4; ++p) {
            float loc = rc + ob[l * 4 + p] * 0.25f * rw * 0.5f;   // off/NPT * w * 0.5
            float x  = loc * (float)Tl - 0.5f;
            float x0 = floorf(x);
            float w  = x - x0;
            int i0 = (int)x0, i1 = i0 + 1;
            float g0 = (i0 >= 0 && i0 < Tl) ? vb[(size_t)i0 * 256] : 0.0f;
            float g1 = (i1 >= 0 && i1 < Tl) ? vb[(size_t)i1 * 256] : 0.0f;
            acc += attp[l * 4 + p] * inv * (g0 * (1.0f - w) + g1 * w);
        }
    }
    samp[(size_t)idx * 256 + h * 128 + ch] = acc;
}

// ---------------------------------------------------------------------------
static void launch_gemm(const float* A, const float* A2, const float* B,
                        const float* bias, const float* mulp, const float* resp,
                        float* C, int M, int N, int K, int relu, hipStream_t stream)
{
    dim3 grid((N + BN_ - 1) / BN_, (M + BM_ - 1) / BM_);
    if (A2)
        gemm_wmma<true,  false, false, false><<<grid, 256, 0, stream>>>(A, A2, B, bias, mulp, resp, C, M, N, K);
    else if (relu)
        gemm_wmma<false, true,  false, false><<<grid, 256, 0, stream>>>(A, A2, B, bias, mulp, resp, C, M, N, K);
    else if (mulp)
        gemm_wmma<false, false, true,  false><<<grid, 256, 0, stream>>>(A, A2, B, bias, mulp, resp, C, M, N, K);
    else if (resp)
        gemm_wmma<false, false, false, true ><<<grid, 256, 0, stream>>>(A, A2, B, bias, mulp, resp, C, M, N, K);
    else
        gemm_wmma<false, false, false, false><<<grid, 256, 0, stream>>>(A, A2, B, bias, mulp, resp, C, M, N, K);
}

extern "C" void kernel_launch(void* const* d_in, const int* in_sizes, int n_in,
                              void* d_out, int out_size, void* d_ws, size_t ws_size,
                              hipStream_t stream)
{
    (void)in_sizes; (void)n_in; (void)out_size; (void)ws_size;

    const float* tgt  = (const float*)d_in[0];
    const float* seg  = (const float*)d_in[1];
    const float* dur  = (const float*)d_in[2];
    const float* src  = (const float*)d_in[3];
    const float* vr   = (const float*)d_in[4];
    const float* so_w = (const float*)d_in[7];  const float* so_b = (const float*)d_in[8];
    const float* aw_w = (const float*)d_in[9];  const float* aw_b = (const float*)d_in[10];
    const float* vp_w = (const float*)d_in[11]; const float* vp_b = (const float*)d_in[12];
    const float* op_w = (const float*)d_in[13]; const float* op_b = (const float*)d_in[14];
    const float* in_w = (const float*)d_in[15]; const float* in_b = (const float*)d_in[16];
    const float* mo_w = (const float*)d_in[17]; const float* mo_b = (const float*)d_in[18];
    const float* n1g  = (const float*)d_in[19]; const float* n1b  = (const float*)d_in[20];
    const float* n2g  = (const float*)d_in[21]; const float* n2b  = (const float*)d_in[22];
    const float* n3g  = (const float*)d_in[23]; const float* n3b  = (const float*)d_in[24];
    const float* l1w  = (const float*)d_in[25]; const float* l1b  = (const float*)d_in[26];
    const float* l2w  = (const float*)d_in[27]; const float* l2b  = (const float*)d_in[28];
    const float* gh_w1 = (const float*)d_in[29]; const float* gh_b1 = (const float*)d_in[30];
    const float* gh_w2 = (const float*)d_in[31]; const float* gh_b2 = (const float*)d_in[32];
    const float* qs_w1 = (const float*)d_in[33]; const float* qs_b1 = (const float*)d_in[34];
    const float* qs_w2 = (const float*)d_in[35]; const float* qs_b2 = (const float*)d_in[36];

    // workspace layout (floats)
    float* ws    = (float*)d_ws;
    float* PE    = ws;                         // NT*512
    float* gate  = PE    + (size_t)NT_ * 512;  // NT*256
    float* tmp   = gate  + (size_t)NT_ * 256;  // NT*1024
    float* qp    = tmp   + (size_t)NT_ * 1024; // NT*256
    float* Qb    = qp    + (size_t)NT_ * 256;
    float* Kb    = Qb    + (size_t)NT_ * 256;
    float* Vb    = Kb    + (size_t)NT_ * 256;
    float* attn  = Vb    + (size_t)NT_ * 256;
    float* samp  = attn  + (size_t)NT_ * 256;
    float* offb  = samp  + (size_t)NT_ * 256;  // NT*32
    float* attLg = offb  + (size_t)NT_ * 32;   // NT*32
    float* rpi   = attLg + (size_t)NT_ * 32;   // NT*8
    float* value = rpi   + (size_t)NT_ * 8;    // BT*256

    float* out = (float*)d_out;
    hipMemcpyAsync(out, tgt, (size_t)NT_ * 256 * sizeof(float),
                   hipMemcpyDeviceToDevice, stream);

    // layer-invariant: pos-embed + gate MLP
    prep_kernel<<<NT_, 256, 0, stream>>>(seg, dur, vr, PE, rpi);
    launch_gemm(PE,  nullptr, gh_w1, gh_b1, nullptr, nullptr, tmp,  NT_, 256, 512, 1, stream);
    launch_gemm(tmp, nullptr, gh_w2, gh_b2, nullptr, nullptr, gate, NT_, 256, 256, 0, stream);

    for (int i = 0; i < L_; ++i) {
        const float* inw0 = in_w + (size_t)(i * 3 + 0) * 65536;
        const float* inw1 = in_w + (size_t)(i * 3 + 1) * 65536;
        const float* inw2 = in_w + (size_t)(i * 3 + 2) * 65536;
        const float* inb0 = in_b + (size_t)(i * 3 + 0) * 256;
        const float* inb1 = in_b + (size_t)(i * 3 + 1) * 256;
        const float* inb2 = in_b + (size_t)(i * 3 + 2) * 256;

        // qp = mlp2(out) * gate
        launch_gemm(out, nullptr, qs_w1, qs_b1, nullptr, nullptr, tmp, NT_, 256, 256, 1, stream);
        launch_gemm(tmp, nullptr, qs_w2, qs_b2, gate,   nullptr, qp,  NT_, 256, 256, 0, stream);

        // self-attention (q = k = out+qp, v = out)
        launch_gemm(out, qp,      inw0, inb0, nullptr, nullptr, Qb, NT_, 256, 256, 0, stream);
        launch_gemm(out, qp,      inw1, inb1, nullptr, nullptr, Kb, NT_, 256, 256, 0, stream);
        launch_gemm(out, nullptr, inw2, inb2, nullptr, nullptr, Vb, NT_, 256, 256, 0, stream);
        mha_flash<<<dim3(57, BS_ * HS_), 32, 0, stream>>>(Qb, Kb, Vb, attn);
        launch_gemm(attn, nullptr, mo_w + (size_t)i * 65536, mo_b + (size_t)i * 256,
                    nullptr, out, out, NT_, 256, 256, 0, stream);
        ln_kernel<<<NT_, 256, 0, stream>>>(out, n2g + i * 256, n2b + i * 256);

        // deformable cross-attention
        launch_gemm(src, nullptr, vp_w + (size_t)i * 65536, vp_b + (size_t)i * 256,
                    nullptr, nullptr, value, BT_, 256, 256, 0, stream);
        launch_gemm(out, qp, so_w + (size_t)i * 256 * 32, so_b + (size_t)i * 32,
                    nullptr, nullptr, offb,  NT_, 32, 256, 0, stream);
        launch_gemm(out, qp, aw_w + (size_t)i * 256 * 32, aw_b + (size_t)i * 32,
                    nullptr, nullptr, attLg, NT_, 32, 256, 0, stream);
        deform_sample<<<NT_, 256, 0, stream>>>(value, offb, attLg, rpi, samp);
        launch_gemm(samp, nullptr, op_w + (size_t)i * 65536, op_b + (size_t)i * 256,
                    nullptr, out, out, NT_, 256, 256, 0, stream);
        ln_kernel<<<NT_, 256, 0, stream>>>(out, n1g + i * 256, n1b + i * 256);

        // FFN
        launch_gemm(out, nullptr, l1w + (size_t)i * 256 * 1024, l1b + (size_t)i * 1024,
                    nullptr, nullptr, tmp, NT_, 1024, 256, 1, stream);
        launch_gemm(tmp, nullptr, l2w + (size_t)i * 1024 * 256, l2b + (size_t)i * 256,
                    nullptr, out, out, NT_, 256, 1024, 0, stream);
        ln_kernel<<<NT_, 256, 0, stream>>>(out, n3g + i * 256, n3b + i * 256);
    }
}
